// LocalHiddenPositiveProjection_31696858645008
// MI455X (gfx1250) — matrix operand order repaired and checked by reading, not compiled, not run on hardware
//
#include <hip/hip_runtime.h>
#include <hip/hip_bf16.h>

// ---- problem constants ----
#define BATCH   2
#define HEADS   6
#define DIM     384
#define NTOK    3136          // 56*56
#define NP      3137          // N + CLS
#define MTILES  (NTOK / 16)   // 196

typedef __bf16 bf16;
typedef __attribute__((ext_vector_type(8)))  __bf16 v8bf;
typedef __attribute__((ext_vector_type(16))) __bf16 v16bf;
typedef __attribute__((ext_vector_type(8)))  float  v8f;

static __device__ __forceinline__ bf16 f2bf(float f) {
    unsigned u = __builtin_bit_cast(unsigned, f);
    unsigned r = (u + 0x7FFFu + ((u >> 16) & 1u)) >> 16;   // round-to-nearest-even
    unsigned short s = (unsigned short)r;
    return __builtin_bit_cast(bf16, s);
}

static __device__ __forceinline__ v16bf join8(v8bf lo, v8bf hi) {
    union { v16bf v; v8bf h[2]; } u;
    u.h[0] = lo; u.h[1] = hi;
    return u.v;
}

// ---------------------------------------------------------------------------
// Kernel 0: fp32 -> bf16 elementwise convert
// ---------------------------------------------------------------------------
__global__ void __launch_bounds__(256) cvt_f32_bf16(const float* __restrict__ src,
                                                    bf16* __restrict__ dst, int n) {
    int i = blockIdx.x * 256 + threadIdx.x;
    if (i < n) dst[i] = f2bf(src[i]);
}

// ---------------------------------------------------------------------------
// Kernel 1: per-row head-mean, min-max normalize, 0.99-quantile clip -> bf16 row
// one workgroup (256 threads) per output row; 6272 blocks total
// ---------------------------------------------------------------------------
__global__ void __launch_bounds__(256) row_kernel(const float* __restrict__ attn,
                                                  bf16* __restrict__ a_out) {
    __shared__ float rowv[NTOK];
    __shared__ float sel[NTOK];
    __shared__ float redv[256];
    __shared__ int   redi[256];
    __shared__ float vals[33];

    const int tid = threadIdx.x;
    const int row = blockIdx.x;          // 0 .. B*N-1
    const int b   = row / NTOK;
    const int n   = row % NTOK;

    const size_t hstride = (size_t)NP * NP;
    const float* base = attn + (size_t)b * HEADS * hstride + (size_t)(n + 1) * NP + 1;

    // head mean + local min/max
    float lmin =  __builtin_inff();
    float lmax = -__builtin_inff();
    for (int m = tid; m < NTOK; m += 256) {
        float s = 0.f;
        #pragma unroll
        for (int h = 0; h < HEADS; ++h) s += base[(size_t)h * hstride + m];
        s *= (1.0f / HEADS);
        rowv[m] = s;
        lmin = fminf(lmin, s);
        lmax = fmaxf(lmax, s);
    }
    __syncthreads();

    // min reduce
    redv[tid] = lmin; __syncthreads();
    for (int s = 128; s > 0; s >>= 1) {
        if (tid < s) redv[tid] = fminf(redv[tid], redv[tid + s]);
        __syncthreads();
    }
    const float amin = redv[0]; __syncthreads();
    // max reduce
    redv[tid] = lmax; __syncthreads();
    for (int s = 128; s > 0; s >>= 1) {
        if (tid < s) redv[tid] = fmaxf(redv[tid], redv[tid + s]);
        __syncthreads();
    }
    const float amax = redv[0]; __syncthreads();

    const float inv = 1.0f / (amax - amin);
    for (int m = tid; m < NTOK; m += 256) {
        float v = (rowv[m] - amin) * inv;
        rowv[m] = v;
        sel[m]  = v;
    }
    __syncthreads();

    // extract top-33 via repeated argmax (values are in [0,1]; -2 marks taken)
    for (int it = 0; it < 33; ++it) {
        float bv = -1.0f; int bi = -1;
        for (int m = tid; m < NTOK; m += 256) {
            float v = sel[m];
            if (v > bv) { bv = v; bi = m; }
        }
        redv[tid] = bv; redi[tid] = bi; __syncthreads();
        for (int s = 128; s > 0; s >>= 1) {
            if (tid < s && redv[tid + s] > redv[tid]) {
                redv[tid] = redv[tid + s];
                redi[tid] = redi[tid + s];
            }
            __syncthreads();
        }
        if (tid == 0) { vals[it] = redv[0]; sel[redi[0]] = -2.0f; }
        __syncthreads();
    }
    // sorted ascending: s[3104] = 32nd largest = vals[31]; s[3103] = 33rd = vals[32]
    // quantile pos = 0.99*(N-1) = 3103.65 -> frac 0.65
    const float q = vals[32] + 0.65f * (vals[31] - vals[32]);

    bf16* dst = a_out + ((size_t)b * NTOK + n) * (size_t)NTOK;
    for (int m = tid; m < NTOK; m += 256) {
        float v = rowv[m];
        dst[m] = f2bf(v > q ? 0.0f : v);
    }
}

// ---------------------------------------------------------------------------
// Kernel 2: mixed[b,n,c] = (1/N) * sum_m a[b,n,m] * code[b,c,m]     (bf16 WMMA)
// workgroup = 16 token rows x 384 channels; 8 waves, 3 c-tiles each; K = 3136
// ---------------------------------------------------------------------------
__global__ void __launch_bounds__(256) gemm_mix(const bf16* __restrict__ A,
                                                const bf16* __restrict__ codeB,
                                                bf16* __restrict__ mixed) {
    const int blk  = blockIdx.x;
    const int mt   = blk % MTILES;
    const int b    = blk / MTILES;
    const int wave = threadIdx.x >> 5;
    const int lane = threadIdx.x & 31;

    const int n0   = mt * 16;
    const int row  = lane & 15;
    const int half = lane >> 4;
    const int c0   = wave * 48;          // 3 tiles of 16 channels per wave

    const bf16* Abase = A + ((size_t)b * NTOK + n0 + row) * (size_t)NTOK + half * 16;
    const bf16* Bbase = codeB + (size_t)b * DIM * NTOK + half * 16;

    v8f acc0 = {}, acc1 = {}, acc2 = {};
    const size_t brow0 = (size_t)(c0 + 0  + row) * NTOK;
    const size_t brow1 = (size_t)(c0 + 16 + row) * NTOK;
    const size_t brow2 = (size_t)(c0 + 32 + row) * NTOK;

    for (int k0 = 0; k0 < NTOK; k0 += 32) {
        v16bf af = join8(*(const v8bf*)(Abase + k0),
                         *(const v8bf*)(Abase + k0 + 8));
        v16bf bf0 = *(const v16bf*)(Bbase + brow0 + k0);
        v16bf bf1 = *(const v16bf*)(Bbase + brow1 + k0);
        v16bf bf2 = *(const v16bf*)(Bbase + brow2 + k0);
        acc0 = __builtin_amdgcn_wmma_f32_16x16x32_bf16(false, af, false, bf0, (short)0, acc0, false, false);
        acc1 = __builtin_amdgcn_wmma_f32_16x16x32_bf16(false, af, false, bf1, (short)0, acc1, false, false);
        acc2 = __builtin_amdgcn_wmma_f32_16x16x32_bf16(false, af, false, bf2, (short)0, acc2, false, false);
    }

    const float invN = 1.0f / (float)NTOK;
    const int hM = half * 8;
    #pragma unroll
    for (int r = 0; r < 8; ++r) {
        const size_t mrow = (size_t)b * NTOK + n0 + hM + r;
        mixed[mrow * DIM + c0 +  0 + row] = f2bf(acc0[r] * invN);
        mixed[mrow * DIM + c0 + 16 + row] = f2bf(acc1[r] * invN);
        mixed[mrow * DIM + c0 + 32 + row] = f2bf(acc2[r] * invN);
    }
}

// ---------------------------------------------------------------------------
// Kernel 3: out = W2 @ relu(W1 @ mixed + b1) + b2   (two bf16 WMMA GEMMs)
// workgroup = 16 tokens; hidden staged in LDS as bf16
// ---------------------------------------------------------------------------
__global__ void __launch_bounds__(256) proj_kernel(const bf16* __restrict__ mixed,
                                                   const bf16* __restrict__ W1b,
                                                   const float* __restrict__ b1,
                                                   const bf16* __restrict__ W2b,
                                                   const float* __restrict__ b2,
                                                   float* __restrict__ out) {
    __shared__ __align__(32) bf16 hbuf[16 * DIM];

    const int blk  = blockIdx.x;
    const int mt   = blk % MTILES;
    const int b    = blk / MTILES;
    const int wave = threadIdx.x >> 5;
    const int lane = threadIdx.x & 31;

    const int n0   = mt * 16;
    const int row  = lane & 15;
    const int half = lane >> 4;
    const int c0   = wave * 48;
    const int hM   = half * 8;

    // ---- GEMM1: h = relu(mixed @ W1^T + b1) ----
    const bf16* Abase = mixed + ((size_t)b * NTOK + n0 + row) * DIM + half * 16;
    v8f acc0 = {}, acc1 = {}, acc2 = {};
    for (int k0 = 0; k0 < DIM; k0 += 32) {
        v16bf af = join8(*(const v8bf*)(Abase + k0),
                         *(const v8bf*)(Abase + k0 + 8));
        v16bf bf0 = *(const v16bf*)(W1b + (size_t)(c0 +  0 + row) * DIM + half * 16 + k0);
        v16bf bf1 = *(const v16bf*)(W1b + (size_t)(c0 + 16 + row) * DIM + half * 16 + k0);
        v16bf bf2 = *(const v16bf*)(W1b + (size_t)(c0 + 32 + row) * DIM + half * 16 + k0);
        acc0 = __builtin_amdgcn_wmma_f32_16x16x32_bf16(false, af, false, bf0, (short)0, acc0, false, false);
        acc1 = __builtin_amdgcn_wmma_f32_16x16x32_bf16(false, af, false, bf1, (short)0, acc1, false, false);
        acc2 = __builtin_amdgcn_wmma_f32_16x16x32_bf16(false, af, false, bf2, (short)0, acc2, false, false);
    }
    #pragma unroll
    for (int r = 0; r < 8; ++r) {
        int c;
        float v;
        c = c0 +  0 + row; v = acc0[r] + b1[c]; hbuf[(hM + r) * DIM + c] = f2bf(fmaxf(v, 0.f));
        c = c0 + 16 + row; v = acc1[r] + b1[c]; hbuf[(hM + r) * DIM + c] = f2bf(fmaxf(v, 0.f));
        c = c0 + 32 + row; v = acc2[r] + b1[c]; hbuf[(hM + r) * DIM + c] = f2bf(fmaxf(v, 0.f));
    }
    __syncthreads();

    // ---- GEMM2: out = h @ W2^T + b2 ----
    const bf16* A2 = hbuf + (size_t)row * DIM + half * 16;
    v8f o0 = {}, o1 = {}, o2 = {};
    for (int k0 = 0; k0 < DIM; k0 += 32) {
        v16bf af = join8(*(const v8bf*)(A2 + k0),
                         *(const v8bf*)(A2 + k0 + 8));
        v16bf bf0 = *(const v16bf*)(W2b + (size_t)(c0 +  0 + row) * DIM + half * 16 + k0);
        v16bf bf1 = *(const v16bf*)(W2b + (size_t)(c0 + 16 + row) * DIM + half * 16 + k0);
        v16bf bf2 = *(const v16bf*)(W2b + (size_t)(c0 + 32 + row) * DIM + half * 16 + k0);
        o0 = __builtin_amdgcn_wmma_f32_16x16x32_bf16(false, af, false, bf0, (short)0, o0, false, false);
        o1 = __builtin_amdgcn_wmma_f32_16x16x32_bf16(false, af, false, bf1, (short)0, o1, false, false);
        o2 = __builtin_amdgcn_wmma_f32_16x16x32_bf16(false, af, false, bf2, (short)0, o2, false, false);
    }
    #pragma unroll
    for (int r = 0; r < 8; ++r) {
        const int n = n0 + hM + r;
        int o;
        o = c0 +  0 + row; out[((size_t)b * DIM + o) * NTOK + n] = o0[r] + b2[o];
        o = c0 + 16 + row; out[((size_t)b * DIM + o) * NTOK + n] = o1[r] + b2[o];
        o = c0 + 32 + row; out[((size_t)b * DIM + o) * NTOK + n] = o2[r] + b2[o];
    }
}

// ---------------------------------------------------------------------------
extern "C" void kernel_launch(void* const* d_in, const int* in_sizes, int n_in,
                              void* d_out, int out_size, void* d_ws, size_t ws_size,
                              hipStream_t stream) {
    const float* code = (const float*)d_in[0];   // [B, DIM, N]
    const float* attn = (const float*)d_in[1];   // [B, HEADS, NP, NP]
    const float* W1   = (const float*)d_in[2];   // [DIM, DIM]
    const float* b1   = (const float*)d_in[3];   // [DIM]
    const float* W2   = (const float*)d_in[4];   // [DIM, DIM]
    const float* b2   = (const float*)d_in[5];   // [DIM]
    float*       out  = (float*)d_out;           // [B, DIM, 56, 56]

    // workspace layout (bytes)
    const size_t code_elems = (size_t)BATCH * DIM * NTOK;      // 2,408,448
    const size_t w_elems    = (size_t)DIM * DIM;               // 147,456
    const size_t a_elems    = (size_t)BATCH * NTOK * NTOK;     // 19,668,992
    const size_t mix_elems  = (size_t)BATCH * NTOK * DIM;

    char* ws = (char*)d_ws;
    bf16* code_bf = (bf16*)ws;                       ws += ((code_elems * 2 + 255) & ~(size_t)255);
    bf16* W1_bf   = (bf16*)ws;                       ws += ((w_elems    * 2 + 255) & ~(size_t)255);
    bf16* W2_bf   = (bf16*)ws;                       ws += ((w_elems    * 2 + 255) & ~(size_t)255);
    bf16* a_bf    = (bf16*)ws;                       ws += ((a_elems    * 2 + 255) & ~(size_t)255);
    bf16* mix_bf  = (bf16*)ws;

    // 0) fp32 -> bf16 converts
    cvt_f32_bf16<<<(int)((code_elems + 255) / 256), 256, 0, stream>>>(code, code_bf, (int)code_elems);
    cvt_f32_bf16<<<(int)((w_elems    + 255) / 256), 256, 0, stream>>>(W1,   W1_bf,   (int)w_elems);
    cvt_f32_bf16<<<(int)((w_elems    + 255) / 256), 256, 0, stream>>>(W2,   W2_bf,   (int)w_elems);

    // 1) row stats + quantile clip -> a_bf [B, N, N]
    row_kernel<<<BATCH * NTOK, 256, 0, stream>>>(attn, a_bf);

    // 2) mixed = (a @ code^T)/N -> bf16 [B, N, DIM]
    gemm_mix<<<BATCH * MTILES, 256, 0, stream>>>(a_bf, code_bf, mix_bf);

    // 3) projection head -> out fp32 [B, DIM, N]
    proj_kernel<<<BATCH * MTILES, 256, 0, stream>>>(mix_bf, W1_bf, b1, W2_bf, b2, out);
}
